// SDPASimple_46797963657669
// MI455X (gfx1250) — matrix operand order repaired and checked by reading, not compiled, not run on hardware
//
#include <hip/hip_runtime.h>
#include <hip/hip_bf16.h>

typedef _Float16 half_t;
typedef __attribute__((ext_vector_type(16))) _Float16 v16h;
typedef __attribute__((ext_vector_type(8)))  _Float16 v8h;
typedef __attribute__((ext_vector_type(4)))  _Float16 v4h;
typedef __attribute__((ext_vector_type(8)))  float    v8f;
typedef __attribute__((ext_vector_type(4)))  float    v4f;
typedef __attribute__((ext_vector_type(4)))  unsigned int v4u;
typedef __attribute__((ext_vector_type(8)))  int      v8i;
typedef __attribute__((ext_vector_type(4)))  int      v4i;

namespace {
constexpr int kDim = 4096, kHeadDim = 128, kBsz = 2, kSeq = 2048;
constexpr int kNH = 32, kNKV = 8;
constexpr int QTILE = 128, KTILE = 64, WAVES = 8;
// LDS row strides in halves: padded to rotate banks, keep 16B alignment
constexpr int SK_STR = 136;   // K rows:   128 + 8  (pad 4 DWORDs / 64 DWORDs)
constexpr int SV_STR = 72;    // V^T rows:  64 + 8  (pad 4 DWORDs / 32 DWORDs)
constexpr int SP_STR = 72;    // P rows:    64 + 8
constexpr float SCALE_LOG2E = 0.12751589167076826f;  // log2(e)/sqrt(128)
}

#if __has_builtin(__builtin_amdgcn_tensor_load_to_lds)
#define HAVE_TDM 1
#else
#define HAVE_TDM 0
#endif

// ds_swizzle XOR-butterfly within each 16-lane group (group-of-32 mode)
#define SWZ_F(v, imm) \
  __int_as_float(__builtin_amdgcn_ds_swizzle(__float_as_int(v), (imm)))

__device__ inline v16h ld16h(const half_t* p) {
  v8h a = *(const v8h*)p;
  v8h b = *(const v8h*)(p + 8);
  return __builtin_shufflevector(a, b, 0,1,2,3,4,5,6,7,8,9,10,11,12,13,14,15);
}

__device__ inline unsigned lds_off(const void* p) {
  return (unsigned)(unsigned long long)p;   // low 32 bits = LDS byte offset
}

#if HAVE_TDM
// Issue one 2-D TDM tile load (f16 elements) with per-row LDS padding.
// D# layout per CDNA5 ISA 8.3/8.4: group0 {count,lds,global,type}, group1 dims.
__device__ inline void tdm_load_2d(unsigned lds_byte, const void* gaddr,
                                   unsigned tensor_d0, unsigned tensor_d1,
                                   unsigned tile_d0, unsigned tile_d1,
                                   unsigned d0_stride,
                                   unsigned pad_interval_code,
                                   unsigned pad_amount_code) {
  const unsigned long long ga = (unsigned long long)gaddr;
  v4u g0;
  g0[0] = 1u;                                         // count=1 (valid), user mode
  g0[1] = lds_byte;                                   // lds_addr (bytes)
  g0[2] = (unsigned)(ga & 0xffffffffu);               // global_addr[31:0]
  g0[3] = (unsigned)((ga >> 32) & 0x1ffffffu) | (2u << 30);  // addr[56:32], type=2
  v8i g1;
  g1[0] = (int)((1u << 16) |                          // data_size = 2 bytes
                (1u << 20) |                          // pad_enable
                (pad_interval_code << 22) |
                (pad_amount_code << 25));
  g1[1] = (int)((tensor_d0 & 0xffffu) << 16);         // tensor_dim0[15:0]
  g1[2] = (int)(((tensor_d0 >> 16) & 0xffffu) |
                ((tensor_d1 & 0xffffu) << 16));       // dim0[31:16] | dim1[15:0]
  g1[3] = (int)(((tensor_d1 >> 16) & 0xffffu) |
                ((tile_d0 & 0xffffu) << 16));         // dim1[31:16] | tile_dim0
  g1[4] = (int)(tile_d1 & 0xffffu);                   // tile_dim1 (tile_dim2=0)
  g1[5] = (int)d0_stride;                             // tensor_dim0_stride[31:0]
  g1[6] = 0;                                          // stride hi / dim1_stride lo
  g1[7] = 0;
  v4i z4 = {};
  v8i z8 = {};
  // 6-arg variant (clang-23 / therock headers): groups 0..3 + extra group + cpol
  __builtin_amdgcn_tensor_load_to_lds(g0, g1, z4, z4, z8, 0);
}
#endif

// Cooperative f16 copy fallback (pre-converted K / V^T tiles)
__device__ inline void stage_copy_f16(half_t* sKb, half_t* sVb,
                                      const half_t* __restrict__ kh,
                                      const half_t* __restrict__ vth, int tid) {
  #pragma unroll
  for (int c = 0; c < 4; ++c) {
    const int idx = tid + c * 256;                     // 0..1023
    const int r  = idx >> 4, col  = (idx & 15) * 8;    // K: 64 x 128 halves
    *(v8h*)(sKb + r * SK_STR + col) = *(const v8h*)(kh + r * kHeadDim + col);
    const int r2 = idx >> 3, col2 = (idx & 7) * 8;     // V^T: 128 x 64 halves
    *(v8h*)(sVb + r2 * SV_STR + col2) =
        *(const v8h*)(vth + (size_t)r2 * kSeq + col2);
  }
}

// ---------------- pre-pass: fp32 -> f16 (Q scaled, V transposed) -------------
__global__ void cvt_qk(const float* __restrict__ src, half_t* __restrict__ dst,
                       int n4, float scale) {
  const int i = blockIdx.x * blockDim.x + threadIdx.x;
  if (i >= n4) return;
  v4f f = ((const v4f*)src)[i];
  v4h h;
  #pragma unroll
  for (int e = 0; e < 4; ++e) h[e] = (half_t)(f[e] * scale);
  ((v4h*)dst)[i] = h;
}

__global__ void cvt_vt(const float* __restrict__ v, half_t* __restrict__ o, int n) {
  const int i = blockIdx.x * blockDim.x + threadIdx.x;
  if (i >= n) return;
  const int d  = i & (kHeadDim - 1);
  const int s  = (i >> 7) & (kSeq - 1);
  const int bh = i >> 18;
  o[((size_t)bh * kHeadDim + d) * kSeq + s] = (half_t)v[i];
}

// ---------------- main flash-attention kernel (TDM double-buffered) ----------
__global__ __launch_bounds__(256)
void sdpa_flash_tdm(const half_t* __restrict__ Qh, const half_t* __restrict__ Kh,
                    const half_t* __restrict__ VTh, float* __restrict__ O) {
  __shared__ half_t sK[2][KTILE * SK_STR];      // [buf][key][d]
  __shared__ half_t sV[2][kHeadDim * SV_STR];   // [buf][d][key]
  __shared__ half_t sP[WAVES * 16 * SP_STR];    // per-wave P 16x64

  const int tid  = threadIdx.x;
  const int lane = tid & 31, wave = tid >> 5;
  const int l16  = lane & 15, hi = lane >> 4;

  const int bh  = blockIdx.x;
  const int b   = bh >> 5, h = bh & 31, hkv = h >> 2;
  const int q0b = blockIdx.y * QTILE;
  const int q0w = q0b + wave * 16;

  const half_t* gQ  = Qh  + ((size_t)(b * kNH  + h)   * kSeq + q0w) * kHeadDim;
  const half_t* gK  = Kh  +  (size_t)(b * kNKV + hkv) * kSeq * kHeadDim;
  const half_t* gVT = VTh +  (size_t)(b * kNKV + hkv) * kHeadDim * kSeq;

  const int nkt    = q0b / KTILE + QTILE / KTILE;
  const int my_nkt = (q0w + 15) / KTILE + 1;

  // ---- stage tile 0 into buffer 0 ----
#if HAVE_TDM
  if (wave == 0) {
    tdm_load_2d(lds_off(&sK[0][0]), gK,  kHeadDim, kSeq, kHeadDim, KTILE,
                kHeadDim, 5u, 3u);                       // 64DW rows, +4DW pad
    tdm_load_2d(lds_off(&sV[0][0]), gVT, kSeq, kHeadDim, KTILE, kHeadDim,
                kSeq, 4u, 3u);                           // 32DW rows, +4DW pad
  }
#else
  stage_copy_f16(&sK[0][0], &sV[0][0], gK, gVT, tid);
#endif

  // ---- Q as 4 A-fragments (already f16, pre-scaled) ----
  v16h aq[4];
  {
    const half_t* qrow = gQ + l16 * kHeadDim + hi * 16;
    #pragma unroll
    for (int j = 0; j < 4; ++j) {
      v8h lo = *(const v8h*)(qrow + 32 * j);
      v8h hh = *(const v8h*)(qrow + 32 * j + 8);
      aq[j] = __builtin_shufflevector(lo, hh,
                                      0,1,2,3,4,5,6,7,8,9,10,11,12,13,14,15);
    }
  }

  float m_r[8], l_r[8];
  v8f o_t[8];
  #pragma unroll
  for (int r = 0; r < 8; ++r) { m_r[r] = -1e30f; l_r[r] = 0.0f; }
  #pragma unroll
  for (int t = 0; t < 8; ++t) { v8f z = {}; o_t[t] = z; }

#if HAVE_TDM
  if (wave == 0) __builtin_amdgcn_s_wait_tensorcnt(0);
#endif
  __syncthreads();

  for (int kt = 0; kt < nkt; ++kt) {
    const int cur = kt & 1;

    // ---- issue stage of tile kt+1 into the other buffer ----
    if (kt + 1 < nkt) {
      const half_t* nk = gK  + (size_t)(kt + 1) * KTILE * kHeadDim;
      const half_t* nv = gVT + (size_t)(kt + 1) * KTILE;
#if HAVE_TDM
      if (wave == 0) {
        tdm_load_2d(lds_off(&sK[cur ^ 1][0]), nk, kHeadDim, kSeq, kHeadDim,
                    KTILE, kHeadDim, 5u, 3u);
        tdm_load_2d(lds_off(&sV[cur ^ 1][0]), nv, kSeq, kHeadDim, KTILE,
                    kHeadDim, kSeq, 4u, 3u);
      }
#else
      stage_copy_f16(&sK[cur ^ 1][0], &sV[cur ^ 1][0], nk, nv, tid);
#endif
    }

    if (kt < my_nkt) {
      const half_t* sKc = &sK[cur][0];
      const half_t* sVc = &sV[cur][0];

      // ---- S = Q K^T : four 16-key subtiles ----
      v8f s[4] = {{}, {}, {}, {}};
      #pragma unroll
      for (int j = 0; j < 4; ++j) {
        #pragma unroll
        for (int sub = 0; sub < 4; ++sub) {
          v16h bk = ld16h(sKc + (sub * 16 + l16) * SK_STR + 32 * j + hi * 16);
          s[sub] = __builtin_amdgcn_wmma_f32_16x16x32_f16(
              false, aq[j], false, bk, (short)0, s[sub], false, false);
        }
      }

      // ---- online softmax; mask only on the diagonal tile ----
      const int  key0       = kt * KTILE;
      const bool needs_mask = (key0 + KTILE - 1) > q0w;
      #pragma unroll
      for (int r = 0; r < 8; ++r) {
        float a[4];
        #pragma unroll
        for (int sub = 0; sub < 4; ++sub) a[sub] = s[sub][r];
        if (needs_mask) {
          const int qg = q0w + r + hi * 8;
          #pragma unroll
          for (int sub = 0; sub < 4; ++sub)
            a[sub] = (key0 + sub * 16 + l16 <= qg) ? a[sub] : -1e30f;
        }
        float mx = fmaxf(fmaxf(a[0], a[1]), fmaxf(a[2], a[3]));
        mx = fmaxf(mx, SWZ_F(mx, 0x201F));
        mx = fmaxf(mx, SWZ_F(mx, 0x101F));
        mx = fmaxf(mx, SWZ_F(mx, 0x081F));
        mx = fmaxf(mx, SWZ_F(mx, 0x041F));
        const float mnew  = fmaxf(m_r[r], mx);
        const float alpha = exp2f(m_r[r] - mnew);
        float p[4];
        #pragma unroll
        for (int sub = 0; sub < 4; ++sub) p[sub] = exp2f(a[sub] - mnew);
        float rs = (p[0] + p[1]) + (p[2] + p[3]);
        rs += SWZ_F(rs, 0x201F);
        rs += SWZ_F(rs, 0x101F);
        rs += SWZ_F(rs, 0x081F);
        rs += SWZ_F(rs, 0x041F);
        l_r[r] = l_r[r] * alpha + rs;
        m_r[r] = mnew;
        #pragma unroll
        for (int t = 0; t < 8; ++t) o_t[t][r] = o_t[t][r] * alpha;
        half_t* pr = sP + (wave * 16 + r + hi * 8) * SP_STR;
        #pragma unroll
        for (int sub = 0; sub < 4; ++sub) pr[sub * 16 + l16] = (half_t)p[sub];
      }

      // ---- O += P V ----
      const half_t* prow = sP + (wave * 16 + l16) * SP_STR;
      v16h pa[2];
      #pragma unroll
      for (int kk = 0; kk < 2; ++kk) {
        v8h plo = *(const v8h*)(prow + kk * 32 + hi * 16);
        v8h phi = *(const v8h*)(prow + kk * 32 + 8 + hi * 16);
        pa[kk] = __builtin_shufflevector(plo, phi,
                                         0,1,2,3,4,5,6,7,8,9,10,11,12,13,14,15);
      }
      #pragma unroll
      for (int t = 0; t < 8; ++t) {
        #pragma unroll
        for (int kk = 0; kk < 2; ++kk) {
          v16h bv = ld16h(sVc + (t * 16 + l16) * SV_STR + kk * 32 + hi * 16);
          o_t[t] = __builtin_amdgcn_wmma_f32_16x16x32_f16(
              false, pa[kk], false, bv, (short)0, o_t[t], false, false);
        }
      }
    }

#if HAVE_TDM
    if (wave == 0) __builtin_amdgcn_s_wait_tensorcnt(0);
#endif
    __syncthreads();
  }

  float* orow = O + (size_t)(b * kSeq + q0w) * kDim + h * kHeadDim;
  #pragma unroll
  for (int r = 0; r < 8; ++r) {
    const float inv = 1.0f / l_r[r];
    float* op = orow + (size_t)(r + hi * 8) * kDim + l16;
    #pragma unroll
    for (int t = 0; t < 8; ++t) op[t * 16] = o_t[t][r] * inv;
  }
}

// ---------------- no-workspace fallback: all-in-one fp32 kernel --------------
__global__ __launch_bounds__(256)
void sdpa_flash_wmma(const float* __restrict__ Q, const float* __restrict__ K,
                     const float* __restrict__ V, float* __restrict__ O) {
  __shared__ half_t sK[KTILE * SK_STR];
  __shared__ half_t sV[kHeadDim * SV_STR];
  __shared__ half_t sP[WAVES * 16 * SP_STR];

  const int tid  = threadIdx.x;
  const int lane = tid & 31, wave = tid >> 5;
  const int l16  = lane & 15, hi = lane >> 4;

  const int bh  = blockIdx.x;
  const int b   = bh >> 5, h = bh & 31, hkv = h >> 2;
  const int q0b = blockIdx.y * QTILE;
  const int q0w = q0b + wave * 16;

  const float* gQ = Q + ((size_t)(b * kNH  + h)   * kSeq + q0w) * kHeadDim;
  const float* gK = K +  (size_t)(b * kNKV + hkv) * kSeq * kHeadDim;
  const float* gV = V +  (size_t)(b * kNKV + hkv) * kSeq * kHeadDim;

  v16h aq[4];
  {
    const float* qrow = gQ + l16 * kHeadDim + hi * 16;
    #pragma unroll
    for (int j = 0; j < 4; ++j) {
      v4f f0 = *(const v4f*)(qrow + 32 * j);
      v4f f1 = *(const v4f*)(qrow + 32 * j + 4);
      v4f f2 = *(const v4f*)(qrow + 32 * j + 8);
      v4f f3 = *(const v4f*)(qrow + 32 * j + 12);
      #pragma unroll
      for (int e = 0; e < 4; ++e) {
        aq[j][e]      = (half_t)(f0[e] * SCALE_LOG2E);
        aq[j][4 + e]  = (half_t)(f1[e] * SCALE_LOG2E);
        aq[j][8 + e]  = (half_t)(f2[e] * SCALE_LOG2E);
        aq[j][12 + e] = (half_t)(f3[e] * SCALE_LOG2E);
      }
    }
  }

  float m_r[8], l_r[8];
  v8f o_t[8];
  #pragma unroll
  for (int r = 0; r < 8; ++r) { m_r[r] = -1e30f; l_r[r] = 0.0f; }
  #pragma unroll
  for (int t = 0; t < 8; ++t) { v8f z = {}; o_t[t] = z; }

  const int nkt    = q0b / KTILE + QTILE / KTILE;
  const int my_nkt = (q0w + 15) / KTILE + 1;

  for (int kt = 0; kt < nkt; ++kt) {
    __syncthreads();
    {
      const int r = tid >> 2;
      const int c = (tid & 3) * 32;
      const float* kp = gK + (size_t)(kt * KTILE + r) * kHeadDim + c;
      const float* vp = gV + (size_t)(kt * KTILE + r) * kHeadDim + c;
      half_t* skr = sK + r * SK_STR + c;
      #pragma unroll
      for (int hf = 0; hf < 2; ++hf) {
        v4f k4[4], w4[4];
        #pragma unroll
        for (int i = 0; i < 4; ++i) {
          k4[i] = *(const v4f*)(kp + hf * 16 + 4 * i);
          w4[i] = *(const v4f*)(vp + hf * 16 + 4 * i);
        }
        #pragma unroll
        for (int i = 0; i < 4; ++i)
          #pragma unroll
          for (int e = 0; e < 4; ++e) {
            const int cc = hf * 16 + 4 * i + e;
            skr[cc] = (half_t)k4[i][e];
            sV[(c + cc) * SV_STR + r] = (half_t)w4[i][e];
          }
      }
      if (kt + 1 < nkt) {
        __builtin_prefetch(kp + KTILE * kHeadDim, 0, 1);
        __builtin_prefetch(vp + KTILE * kHeadDim, 0, 1);
      }
    }
    __syncthreads();
    if (kt >= my_nkt) continue;

    v8f s[4] = {{}, {}, {}, {}};
    #pragma unroll
    for (int j = 0; j < 4; ++j) {
      #pragma unroll
      for (int sub = 0; sub < 4; ++sub) {
        v16h bk = ld16h(sK + (sub * 16 + l16) * SK_STR + 32 * j + hi * 16);
        s[sub] = __builtin_amdgcn_wmma_f32_16x16x32_f16(
            false, aq[j], false, bk, (short)0, s[sub], false, false);
      }
    }

    const int  key0       = kt * KTILE;
    const bool needs_mask = (key0 + KTILE - 1) > q0w;
    #pragma unroll
    for (int r = 0; r < 8; ++r) {
      float a[4];
      #pragma unroll
      for (int sub = 0; sub < 4; ++sub) a[sub] = s[sub][r];
      if (needs_mask) {
        const int qg = q0w + r + hi * 8;
        #pragma unroll
        for (int sub = 0; sub < 4; ++sub)
          a[sub] = (key0 + sub * 16 + l16 <= qg) ? a[sub] : -1e30f;
      }
      float mx = fmaxf(fmaxf(a[0], a[1]), fmaxf(a[2], a[3]));
      mx = fmaxf(mx, SWZ_F(mx, 0x201F));
      mx = fmaxf(mx, SWZ_F(mx, 0x101F));
      mx = fmaxf(mx, SWZ_F(mx, 0x081F));
      mx = fmaxf(mx, SWZ_F(mx, 0x041F));
      const float mnew  = fmaxf(m_r[r], mx);
      const float alpha = exp2f(m_r[r] - mnew);
      float p[4];
      #pragma unroll
      for (int sub = 0; sub < 4; ++sub) p[sub] = exp2f(a[sub] - mnew);
      float rs = (p[0] + p[1]) + (p[2] + p[3]);
      rs += SWZ_F(rs, 0x201F);
      rs += SWZ_F(rs, 0x101F);
      rs += SWZ_F(rs, 0x081F);
      rs += SWZ_F(rs, 0x041F);
      l_r[r] = l_r[r] * alpha + rs;
      m_r[r] = mnew;
      #pragma unroll
      for (int t = 0; t < 8; ++t) o_t[t][r] = o_t[t][r] * alpha;
      half_t* pr = sP + (wave * 16 + r + hi * 8) * SP_STR;
      #pragma unroll
      for (int sub = 0; sub < 4; ++sub) pr[sub * 16 + l16] = (half_t)p[sub];
    }

    const half_t* prow = sP + (wave * 16 + l16) * SP_STR;
    v16h pa[2];
    #pragma unroll
    for (int kk = 0; kk < 2; ++kk) {
      v8h plo = *(const v8h*)(prow + kk * 32 + hi * 16);
      v8h phi = *(const v8h*)(prow + kk * 32 + 8 + hi * 16);
      pa[kk] = __builtin_shufflevector(plo, phi,
                                       0,1,2,3,4,5,6,7,8,9,10,11,12,13,14,15);
    }
    #pragma unroll
    for (int t = 0; t < 8; ++t) {
      #pragma unroll
      for (int kk = 0; kk < 2; ++kk) {
        v16h bv = ld16h(sV + (t * 16 + l16) * SV_STR + kk * 32 + hi * 16);
        o_t[t] = __builtin_amdgcn_wmma_f32_16x16x32_f16(
            false, pa[kk], false, bv, (short)0, o_t[t], false, false);
      }
    }
  }

  float* orow = O + (size_t)(b * kSeq + q0w) * kDim + h * kHeadDim;
  #pragma unroll
  for (int r = 0; r < 8; ++r) {
    const float inv = 1.0f / l_r[r];
    float* op = orow + (size_t)(r + hi * 8) * kDim + l16;
    #pragma unroll
    for (int t = 0; t < 8; ++t) op[t * 16] = o_t[t][r] * inv;
  }
}

extern "C" void kernel_launch(void* const* d_in, const int* in_sizes, int n_in,
                              void* d_out, int out_size, void* d_ws, size_t ws_size,
                              hipStream_t stream) {
  // setup_inputs order: 0 input_pos(i64), 1 q, 2 k, 3 v, 4 bsz, 5 seqlen, 6 mask
  const float* q = (const float*)d_in[1];
  const float* k = (const float*)d_in[2];
  const float* v = (const float*)d_in[3];
  float* out = (float*)d_out;
  dim3 grid(kBsz * kNH, kSeq / QTILE);

  const size_t nQ  = (size_t)kBsz * kNH  * kSeq * kHeadDim;   // 16.7M elems
  const size_t nKV = (size_t)kBsz * kNKV * kSeq * kHeadDim;   //  4.2M elems
  const size_t need = (nQ + 2 * nKV) * sizeof(half_t);        //  48 MB

  if (ws_size >= need) {
    half_t* Qh  = (half_t*)d_ws;
    half_t* Kh  = Qh + nQ;
    half_t* VTh = Kh + nKV;
    cvt_qk<<<(int)((nQ / 4 + 255) / 256), 256, 0, stream>>>(q, Qh, (int)(nQ / 4),
                                                            SCALE_LOG2E);
    cvt_qk<<<(int)((nKV / 4 + 255) / 256), 256, 0, stream>>>(k, Kh, (int)(nKV / 4),
                                                             1.0f);
    cvt_vt<<<(int)((nKV + 255) / 256), 256, 0, stream>>>(v, VTh, (int)nKV);
    sdpa_flash_tdm<<<grid, dim3(256), 0, stream>>>(Qh, Kh, VTh, out);
  } else {
    sdpa_flash_wmma<<<grid, dim3(256), 0, stream>>>(q, k, v, out);
  }
}
